// NeuroPhysioBicameral_3934190043679
// MI455X (gfx1250) — matrix operand order repaired and checked by reading, not compiled, not run on hardware
//
#include <hip/hip_runtime.h>
#include <hip/hip_bf16.h>
#include <math.h>

// ---------------------------------------------------------------------------
// Problem constants (match reference)
// ---------------------------------------------------------------------------
#define BB    16384
#define NN    64
#define DD    64
#define HH    16
#define VOCAB 1024
#define INF   64
#define ND    4096   // N*D

typedef __attribute__((ext_vector_type(16))) __bf16 v16bf;
typedef __attribute__((ext_vector_type(8)))  __bf16 v8bf;
typedef __attribute__((ext_vector_type(8)))  float  v8f;

// ---------------------------------------------------------------------------
// WMMA helpers (gfx1250, wave32): D(16x16,f32) += A(16x32,bf16) * B(32x16,bf16)
// ---------------------------------------------------------------------------
__device__ __forceinline__ v8f wmma_bf16(v16bf a, v16bf b, v8f c) {
  return __builtin_amdgcn_wmma_f32_16x16x32_bf16(
      /*neg_a=*/false, a, /*neg_b=*/false, b,
      /*c_mod=*/(short)0, c, /*reuse_a=*/false, /*reuse_b=*/false);
}

// A fragment, row-major source (16 rows x 32 k), ld in elements.
// ISA layout: lanes 0-15 -> K 0..7 & 16..23 ; lanes 16-31 -> K 8..15 & 24..31
__device__ __forceinline__ v16bf frag_a_rm(const __bf16* base, int ld) {
  const int lane = threadIdx.x & 31;
  const __bf16* p = base + (lane & 15) * ld + ((lane >> 4) * 8);
  union { v16bf v; v8bf h[2]; } u;
  u.h[0] = *(const v8bf*)p;
  u.h[1] = *(const v8bf*)(p + 16);
  return u.v;
}

// B fragment from "Bt" storage (rows = output col n, cols = k, row-major).
// ISA layout: lanes 0-15 -> K 0..7 & 16..23 ; lanes 16-31 -> K 8..15 & 24..31, col = lane&15
__device__ __forceinline__ v16bf frag_b_nt(const __bf16* base, int ld) {
  const int lane = threadIdx.x & 31;
  const __bf16* p = base + (lane & 15) * ld + ((lane >> 4) * 8);
  union { v16bf v; v8bf h[2]; } u;
  u.h[0] = *(const v8bf*)p;
  u.h[1] = *(const v8bf*)(p + 16);
  return u.v;
}

// ---------------------------------------------------------------------------
// CDNA5 async global->LDS (16B per lane, tracked by ASYNCcnt)
// ---------------------------------------------------------------------------
__device__ __forceinline__ unsigned lds_addr_u32(const void* p) {
  typedef __attribute__((address_space(3))) const unsigned char as3_byte;
  return (unsigned)(unsigned long long)(as3_byte*)p;
}

__device__ __forceinline__ void async_copy_b128(void* lds_dst,
                                                const void* gsrc) {
  unsigned loff = lds_addr_u32(lds_dst);
  asm volatile("global_load_async_to_lds_b128 %0, %1, off"
               :
               : "v"(loff), "v"((unsigned long long)gsrc)
               : "memory");
}

__device__ __forceinline__ void wait_async_zero() {
  asm volatile("s_wait_asynccnt 0x0" ::: "memory");
}

// ---------------------------------------------------------------------------
// Utility kernels
// ---------------------------------------------------------------------------
__global__ void cvt_bf16_kernel(const float* __restrict__ s,
                                __bf16* __restrict__ d, size_t n) {
  size_t i  = (size_t)blockIdx.x * blockDim.x + threadIdx.x;
  size_t st = (size_t)gridDim.x * blockDim.x;
  for (; i < n; i += st) d[i] = (__bf16)s[i];
}

__global__ void zero_f32_kernel(float* __restrict__ p, int n) {
  int i = blockIdx.x * blockDim.x + threadIdx.x;
  if (i < n) p[i] = 0.f;
}

__global__ __launch_bounds__(256)
void wnorm_kernel(const float* __restrict__ W_slow, float* __restrict__ wnorm) {
  const int n = blockIdx.x;
  float s = 0.f;
  for (int i = threadIdx.x; i < DD * DD; i += 256) {
    float v = W_slow[(size_t)n * DD * DD + i];
    s += v * v;
  }
  __shared__ float red[256];
  red[threadIdx.x] = s;
  __syncthreads();
  for (int st = 128; st > 0; st >>= 1) {
    if (threadIdx.x < st) red[threadIdx.x] += red[threadIdx.x + st];
    __syncthreads();
  }
  if (threadIdx.x == 0) wnorm[n] = sqrtf(red[0]);
}

// ---------------------------------------------------------------------------
// Phase 1: per (n, 256-row block):
//   h = x @ W_enc_n^T + b_enc   (WMMA)
//   slow = h @ W_slow_n^T       (WMMA)
//   hebb_n += slow^T @ h        (WMMA partial, f32 atomics)
//   stats -> regulator MLP -> gate/beta, rate & phys sums
// ---------------------------------------------------------------------------
__global__ __launch_bounds__(256)
void phase1_kernel(const __bf16* __restrict__ xb,
                   const __bf16* __restrict__ wencb,
                   const __bf16* __restrict__ wslowb,
                   const float*  __restrict__ b_enc,
                   const float*  __restrict__ reg_w1,
                   const float*  __restrict__ reg_b1,
                   const float*  __restrict__ reg_g1,
                   const float*  __restrict__ reg_be1,
                   const float*  __restrict__ reg_w2,
                   const float*  __restrict__ reg_b2,
                   const float*  __restrict__ wnorm,
                   __bf16* __restrict__ hb,
                   __bf16* __restrict__ slowb,
                   float* __restrict__ gate_ws,
                   float* __restrict__ beta_ws,
                   float* __restrict__ hebb_ws,
                   float* __restrict__ rate_acc,
                   float* __restrict__ phys_acc) {
  __shared__ __bf16 wenc_s[DD * INF];   // Bt layout: row d, col k
  __shared__ __bf16 wslow_s[DD * DD];   // Bt layout: row o, col d
  __shared__ float  h_s[64 * DD];
  __shared__ __bf16 hbf_s[64 * DD];
  __shared__ float  slow_s[64 * DD];

  const int n    = blockIdx.y;
  const int tid  = threadIdx.x;
  const int w    = tid >> 5;
  const int lane = tid & 31;

  // Stage per-n weights once (async global->LDS, 16B chunks)
  for (int i = tid; i < (DD * INF) / 8; i += 256) {
    async_copy_b128(&((v8bf*)wenc_s)[i],
                    &((const v8bf*)(wencb + (size_t)n * DD * INF))[i]);
    async_copy_b128(&((v8bf*)wslow_s)[i],
                    &((const v8bf*)(wslowb + (size_t)n * DD * DD))[i]);
  }
  wait_async_zero();
  __syncthreads();

  const float wn = wnorm[n];
  v8f hacc[2];
  { v8f z = {}; hacc[0] = z; hacc[1] = z; }
  float metab_sum = 0.f, ps0 = 0.f, ps1 = 0.f, ps2 = 0.f;

  for (int sub = 0; sub < 4; ++sub) {
    const int r0 = (blockIdx.x * 4 + sub) * 64;

    // ---- h = x @ W_enc^T + b (64 x 64 tile, 16 WMMA tiles, 2 per wave) ----
#pragma unroll
    for (int t = 0; t < 2; ++t) {
      const int tt = 2 * w + t, mt = tt >> 2, ct = tt & 3;
      v8f acc = {};
#pragma unroll
      for (int k0 = 0; k0 < INF; k0 += 32) {
        v16bf a = frag_a_rm(xb + (size_t)(r0 + mt * 16) * INF + k0, INF);
        v16bf b = frag_b_nt(wenc_s + (ct * 16) * INF + k0, INF);
        acc = wmma_bf16(a, b, acc);
      }
      const float bias = b_enc[n * DD + ct * 16 + (lane & 15)];
#pragma unroll
      for (int i = 0; i < 8; ++i) {
        const int rr = mt * 16 + (lane >> 4) * 8 + i;
        const int cc = ct * 16 + (lane & 15);
        float v = acc[i] + bias;
        h_s[rr * DD + cc] = v;
        __bf16 vb = (__bf16)v;
        hbf_s[rr * DD + cc] = vb;
        hb[(size_t)(r0 + rr) * ND + n * DD + cc] = vb;
      }
    }
    __syncthreads();

    // ---- slow = h @ W_slow^T ----
#pragma unroll
    for (int t = 0; t < 2; ++t) {
      const int tt = 2 * w + t, mt = tt >> 2, ct = tt & 3;
      v8f acc = {};
#pragma unroll
      for (int k0 = 0; k0 < DD; k0 += 32) {
        v16bf a = frag_a_rm(hbf_s + (mt * 16) * DD + k0, DD);
        v16bf b = frag_b_nt(wslow_s + (ct * 16) * DD + k0, DD);
        acc = wmma_bf16(a, b, acc);
      }
#pragma unroll
      for (int i = 0; i < 8; ++i) {
        const int rr = mt * 16 + (lane >> 4) * 8 + i;
        const int cc = ct * 16 + (lane & 15);
        slow_s[rr * DD + cc] = acc[i];
        slowb[(size_t)(r0 + rr) * ND + n * DD + cc] = (__bf16)acc[i];
      }
    }
    __syncthreads();

    // ---- hebb partial: (o,d) += slow^T @ h over these 64 rows (WMMA) ----
#pragma unroll
    for (int t = 0; t < 2; ++t) {
      const int tt = 2 * w + t, ot = tt >> 2, dt = tt & 3;
#pragma unroll
      for (int k0 = 0; k0 < 64; k0 += 32) {
        union { v16bf v; __bf16 e[16]; } ua;   // A(m=o, k=row) = slow[row, o]
        const int m = lane & 15, kb = (lane >> 4) * 16;
#pragma unroll
        for (int i = 0; i < 8; ++i) {
          ua.e[i]     = (__bf16)slow_s[(k0 + kb + i)      * DD + ot * 16 + m];
          ua.e[8 + i] = (__bf16)slow_s[(k0 + kb + 8 + i) * DD + ot * 16 + m];
        }
        union { v16bf v; __bf16 e[16]; } ub;   // B(k=row, n=d) = h[row, d]
        const int nc = lane & 15, kb2 = (lane >> 4) * 16;
#pragma unroll
        for (int i = 0; i < 16; ++i)
          ub.e[i] = hbf_s[(k0 + kb2 + i) * DD + dt * 16 + nc];
        hacc[t] = wmma_bf16(ua.v, ub.v, hacc[t]);
      }
    }

    // ---- per-row stats + regulator MLP (one thread per row) ----
    if (tid < 64) {
      const int row = r0 + tid;
      float s = 0.f;
      for (int d = 0; d < DD; ++d) s += h_s[tid * DD + d];
      const float mu = s * (1.f / DD);
      float v = 0.f, e = 0.f;
      for (int d = 0; d < DD; ++d) {
        float dv = h_s[tid * DD + d] - mu;
        v += dv * dv;
        e += fabsf(slow_s[tid * DD + d]);
      }
      const float stress = v * (1.f / (DD - 1));   // ddof=1
      const float excit  = e * (1.f / DD);
      const float f0 = stress, f1 = excit, f2 = wn, f3 = 0.5f;

      float h1[HH];
      float mu1 = 0.f;
#pragma unroll
      for (int hh = 0; hh < HH; ++hh) {
        const float* wp = reg_w1 + (size_t)n * HH * 4 + hh * 4;
        float a = wp[0] * f0 + wp[1] * f1 + wp[2] * f2 + wp[3] * f3
                + reg_b1[n * HH + hh];
        h1[hh] = a;
        mu1 += a;
      }
      mu1 *= (1.f / HH);
      float var1 = 0.f;
#pragma unroll
      for (int hh = 0; hh < HH; ++hh) {
        float dv = h1[hh] - mu1;
        var1 += dv * dv;
      }
      var1 *= (1.f / HH);
      const float rs = rsqrtf(var1 + 1e-5f);
      float t1v[HH];
#pragma unroll
      for (int hh = 0; hh < HH; ++hh)
        t1v[hh] = tanhf((h1[hh] - mu1) * rs * reg_g1[n * HH + hh]
                        + reg_be1[n * HH + hh]);
      float ph[3];
#pragma unroll
      for (int o = 0; o < 3; ++o) {
        float a = reg_b2[n * 3 + o];
#pragma unroll
        for (int hh = 0; hh < HH; ++hh)
          a += t1v[hh] * reg_w2[(size_t)n * 3 * HH + o * HH + hh];
        ph[o] = 1.f / (1.f + expf(-a));
      }
      gate_ws[(size_t)row * NN + n] = ph[2];
      beta_ws[(size_t)row * NN + n] = 0.5f + 2.f * ph[1];
      metab_sum += ph[0];
      ps0 += ph[0]; ps1 += ph[1]; ps2 += ph[2];
    }
    __syncthreads();
  }

  // ---- reduce hebb partials & scalar sums to global (f32 atomics) ----
#pragma unroll
  for (int t = 0; t < 2; ++t) {
    const int tt = 2 * w + t, ot = tt >> 2, dt = tt & 3;
#pragma unroll
    for (int i = 0; i < 8; ++i) {
      const int o = ot * 16 + (lane >> 4) * 8 + i;
      const int d = dt * 16 + (lane & 15);
      atomicAdd(&hebb_ws[(size_t)n * DD * DD + o * DD + d], hacc[t][i]);
    }
  }
  if (tid < 64) {
    atomicAdd(&rate_acc[n], metab_sum);
    atomicAdd(&phys_acc[0], ps0);
    atomicAdd(&phys_acc[1], ps1);
    atomicAdd(&phys_acc[2], ps2);
  }
}

// ---------------------------------------------------------------------------
// W_fast_new = (W_fast + rate * tanh(hebb/B)) * 0.9995, store bf16 (Bt layout)
// ---------------------------------------------------------------------------
__global__ __launch_bounds__(256)
void wfast_kernel(const float* __restrict__ W_fast,
                  const float* __restrict__ hebb_ws,
                  const float* __restrict__ rate_acc,
                  __bf16* __restrict__ wfastb) {
  const int n = blockIdx.x;
  const float rate = rate_acc[n] * (0.1f / (float)BB);
  for (int i = threadIdx.x; i < DD * DD; i += 256) {
    const size_t idx = (size_t)n * DD * DD + i;
    float hv = hebb_ws[idx] * (1.f / (float)BB);
    float wv = (W_fast[idx] + rate * tanhf(hv)) * 0.9995f;
    wfastb[idx] = (__bf16)wv;
  }
}

// ---------------------------------------------------------------------------
// Phase 2: fast = h @ W_fast_new^T (WMMA); combined/swish; LayerNorm -> flat
// ---------------------------------------------------------------------------
__global__ __launch_bounds__(256)
void phase2_kernel(const __bf16* __restrict__ hb,
                   const __bf16* __restrict__ slowb,
                   const __bf16* __restrict__ wfastb,
                   const float*  __restrict__ gate_ws,
                   const float*  __restrict__ beta_ws,
                   const float*  __restrict__ ln_g,
                   const float*  __restrict__ ln_b,
                   __bf16* __restrict__ flatb) {
  __shared__ __bf16 wf_s[DD * DD];
  __shared__ float  out_s[64 * DD];

  const int n    = blockIdx.y;
  const int r0   = blockIdx.x * 64;
  const int tid  = threadIdx.x;
  const int w    = tid >> 5;
  const int lane = tid & 31;

  for (int i = tid; i < (DD * DD) / 8; i += 256)
    async_copy_b128(&((v8bf*)wf_s)[i],
                    &((const v8bf*)(wfastb + (size_t)n * DD * DD))[i]);
  wait_async_zero();
  __syncthreads();

#pragma unroll
  for (int t = 0; t < 2; ++t) {
    const int tt = 2 * w + t, mt = tt >> 2, ct = tt & 3;
    v8f acc = {};
#pragma unroll
    for (int k0 = 0; k0 < DD; k0 += 32) {
      v16bf a = frag_a_rm(hb + (size_t)(r0 + mt * 16) * ND + n * DD + k0, ND);
      v16bf b = frag_b_nt(wf_s + (ct * 16) * DD + k0, DD);
      acc = wmma_bf16(a, b, acc);
    }
#pragma unroll
    for (int i = 0; i < 8; ++i) {
      const int rr  = mt * 16 + (lane >> 4) * 8 + i;
      const int cc  = ct * 16 + (lane & 15);
      const int row = r0 + rr;
      float fast = acc[i];
      float sl   = (float)slowb[(size_t)row * ND + n * DD + cc];
      float g    = gate_ws[(size_t)row * NN + n];
      float bt   = beta_ws[(size_t)row * NN + n];
      float comb = sl + g * fast;
      out_s[rr * DD + cc] = comb / (1.f + expf(-bt * comb));  // swish(beta)
    }
  }
  __syncthreads();

  if (tid < 64) {
    float s = 0.f;
    for (int d = 0; d < DD; ++d) s += out_s[tid * DD + d];
    const float mu = s * (1.f / DD);
    float v = 0.f;
    for (int d = 0; d < DD; ++d) {
      float dv = out_s[tid * DD + d] - mu;
      v += dv * dv;
    }
    const float rs = rsqrtf(v * (1.f / DD) + 1e-5f);
    const int row = r0 + tid;
    for (int d = 0; d < DD; ++d) {
      float y = (out_s[tid * DD + d] - mu) * rs * ln_g[n * DD + d]
              + ln_b[n * DD + d];
      flatb[(size_t)row * ND + n * DD + d] = (__bf16)y;
    }
  }
}

// ---------------------------------------------------------------------------
// Big GEMM: C[M x Nc] = A(M x K) @ Bt(Nc x K)^T + bias, bf16 in, bf16/f32 out
// 128x128 block tile, 8 waves -> 64x32 per wave, K-step 32.
// Double-buffered LDS staging via CDNA5 async global->LDS (ASYNCcnt).
// ---------------------------------------------------------------------------
__global__ __launch_bounds__(256)
void gemm_kernel(const __bf16* __restrict__ A,
                 const __bf16* __restrict__ Bt,
                 const float*  __restrict__ bias,
                 void* __restrict__ Cout,
                 int M, int Nc, int K, int out_is_bf16) {
  __shared__ __bf16 As[2][128 * 32];
  __shared__ __bf16 Bs[2][128 * 32];

  const int m0   = blockIdx.x * 128;
  const int n0   = blockIdx.y * 128;
  const int tid  = threadIdx.x;
  const int w    = tid >> 5;
  const int lane = tid & 31;
  const int wm   = w & 1;        // 2 wave-rows
  const int wn   = w >> 1;       // 4 wave-cols

  v8f acc[4][2];
  {
    v8f z = {};
#pragma unroll
    for (int ms = 0; ms < 4; ++ms)
#pragma unroll
      for (int ns = 0; ns < 2; ++ns) acc[ms][ns] = z;
  }

  // issue async copies for one 128x32 A tile + 128x32 Bt tile
  auto stage = [&](int buf, int kt) {
    const int k0 = kt * 32;
#pragma unroll
    for (int j = 0; j < 2; ++j) {
      const int c   = tid + 256 * j;  // 512 16B-chunks per tile
      const int row = c >> 2;
      const int off = (c & 3) * 8;
      async_copy_b128(&As[buf][row * 32 + off],
                      &A[(size_t)(m0 + row) * K + k0 + off]);
      async_copy_b128(&Bs[buf][row * 32 + off],
                      &Bt[(size_t)(n0 + row) * K + k0 + off]);
    }
  };

  const int KT = K / 32;
  stage(0, 0);
  int buf = 0;
  for (int kt = 0; kt < KT; ++kt) {
    wait_async_zero();   // our async writes into `buf` have landed in LDS
    __syncthreads();     // everyone's writes visible; prev reads of buf^1 done
    if (kt + 1 < KT) stage(buf ^ 1, kt + 1);

    v16bf bfr[2];
#pragma unroll
    for (int ns = 0; ns < 2; ++ns)
      bfr[ns] = frag_b_nt(&Bs[buf][(wn * 32 + ns * 16) * 32], 32);
#pragma unroll
    for (int ms = 0; ms < 4; ++ms) {
      v16bf afr = frag_a_rm(&As[buf][(wm * 64 + ms * 16) * 32], 32);
#pragma unroll
      for (int ns = 0; ns < 2; ++ns)
        acc[ms][ns] = wmma_bf16(afr, bfr[ns], acc[ms][ns]);
    }
    buf ^= 1;
  }

  // epilogue
#pragma unroll
  for (int ms = 0; ms < 4; ++ms) {
#pragma unroll
    for (int ns = 0; ns < 2; ++ns) {
#pragma unroll
      for (int i = 0; i < 8; ++i) {
        const int row = m0 + wm * 64 + ms * 16 + (lane >> 4) * 8 + i;
        const int col = n0 + wn * 32 + ns * 16 + (lane & 15);
        float v = acc[ms][ns][i] + bias[col];
        if (out_is_bf16)
          ((__bf16*)Cout)[(size_t)row * Nc + col] = (__bf16)v;
        else
          ((float*)Cout)[(size_t)row * Nc + col] = v;
      }
    }
  }
}

__global__ void finalize_phys_kernel(const float* __restrict__ phys_acc,
                                     float* __restrict__ out) {
  if (threadIdx.x < 3)
    out[(size_t)BB * VOCAB + threadIdx.x] =
        phys_acc[threadIdx.x] * (1.f / ((float)BB * (float)NN));
}

// ---------------------------------------------------------------------------
// Host launch
// ---------------------------------------------------------------------------
extern "C" void kernel_launch(void* const* d_in, const int* in_sizes, int n_in,
                              void* d_out, int out_size, void* d_ws,
                              size_t ws_size, hipStream_t stream) {
  const float* x       = (const float*)d_in[0];
  const float* W_enc   = (const float*)d_in[1];
  const float* b_enc   = (const float*)d_in[2];
  const float* W_slow  = (const float*)d_in[3];
  const float* W_fast  = (const float*)d_in[4];
  const float* reg_w1  = (const float*)d_in[5];
  const float* reg_b1  = (const float*)d_in[6];
  const float* reg_g1  = (const float*)d_in[7];
  const float* reg_be1 = (const float*)d_in[8];
  const float* reg_w2  = (const float*)d_in[9];
  const float* reg_b2  = (const float*)d_in[10];
  const float* ln_g    = (const float*)d_in[11];
  const float* ln_b    = (const float*)d_in[12];
  const float* W_cc    = (const float*)d_in[13];
  const float* b_cc    = (const float*)d_in[14];
  const float* W_ro    = (const float*)d_in[15];
  const float* b_ro    = (const float*)d_in[16];
  float* out = (float*)d_out;

  // workspace carve-up
  char* base = (char*)d_ws;
  size_t off = 0;
  auto carve = [&](size_t bytes) -> char* {
    char* p = base + off;
    off = (off + bytes + 255) & ~(size_t)255;
    return p;
  };
  __bf16* xb     = (__bf16*)carve((size_t)BB * INF * 2);
  __bf16* wencb  = (__bf16*)carve((size_t)ND * INF * 2);
  __bf16* wslowb = (__bf16*)carve((size_t)NN * DD * DD * 2);
  __bf16* wfastb = (__bf16*)carve((size_t)NN * DD * DD * 2);
  __bf16* wccb   = (__bf16*)carve((size_t)ND * ND * 2);
  __bf16* wrob   = (__bf16*)carve((size_t)VOCAB * ND * 2);
  __bf16* hb     = (__bf16*)carve((size_t)BB * ND * 2);
  __bf16* slowb  = (__bf16*)carve((size_t)BB * ND * 2);
  __bf16* flatb  = (__bf16*)carve((size_t)BB * ND * 2);
  __bf16* calb   = (__bf16*)carve((size_t)BB * ND * 2);
  float*  gate_w = (float*)carve((size_t)BB * NN * 4);
  float*  beta_w = (float*)carve((size_t)BB * NN * 4);
  float*  hebb_w = (float*)carve((size_t)NN * DD * DD * 4);
  float*  wnormw = (float*)carve((size_t)NN * 4);
  float*  ratew  = (float*)carve((size_t)NN * 4);
  float*  physw  = (float*)carve((size_t)4 * 4);

  // 1) f32 -> bf16 packs
  cvt_bf16_kernel<<<2048, 256, 0, stream>>>(x,      xb,     (size_t)BB * INF);
  cvt_bf16_kernel<<<1024, 256, 0, stream>>>(W_enc,  wencb,  (size_t)ND * INF);
  cvt_bf16_kernel<<<1024, 256, 0, stream>>>(W_slow, wslowb, (size_t)NN * DD * DD);
  cvt_bf16_kernel<<<4096, 256, 0, stream>>>(W_cc,   wccb,   (size_t)ND * ND);
  cvt_bf16_kernel<<<2048, 256, 0, stream>>>(W_ro,   wrob,   (size_t)VOCAB * ND);

  // 2) zero accumulators (every call: deterministic)
  zero_f32_kernel<<<(NN * DD * DD + 255) / 256, 256, 0, stream>>>(hebb_w,
                                                                  NN * DD * DD);
  zero_f32_kernel<<<1, 256, 0, stream>>>(ratew, NN);
  zero_f32_kernel<<<1, 256, 0, stream>>>(physw, 4);

  // 3) w_norm
  wnorm_kernel<<<NN, 256, 0, stream>>>(W_slow, wnormw);

  // 4) phase 1: h, slow, hebb, gates
  phase1_kernel<<<dim3(BB / 256, NN), 256, 0, stream>>>(
      xb, wencb, wslowb, b_enc, reg_w1, reg_b1, reg_g1, reg_be1, reg_w2,
      reg_b2, wnormw, hb, slowb, gate_w, beta_w, hebb_w, ratew, physw);

  // 5) W_fast_new
  wfast_kernel<<<NN, 256, 0, stream>>>(W_fast, hebb_w, ratew, wfastb);

  // 6) phase 2: fast + swish + LN -> flat
  phase2_kernel<<<dim3(BB / 64, NN), 256, 0, stream>>>(
      hb, slowb, wfastb, gate_w, beta_w, ln_g, ln_b, flatb);

  // 7) callosal = flat @ W_cc^T + b_cc  (bf16 out)
  gemm_kernel<<<dim3(BB / 128, ND / 128), 256, 0, stream>>>(
      flatb, wccb, b_cc, (void*)calb, BB, ND, ND, 1);

  // 8) logits = callosal @ W_ro^T + b_ro  (f32 out -> d_out)
  gemm_kernel<<<dim3(BB / 128, VOCAB / 128), 256, 0, stream>>>(
      calb, wrob, b_ro, (void*)out, BB, VOCAB, ND, 0);

  // 9) phys_avg
  finalize_phys_kernel<<<1, 32, 0, stream>>>(physw, out);
}
